// TransposeDISCOBlock_35210141893037
// MI455X (gfx1250) — matrix operand order repaired and checked by reading, not compiled, not run on hardware
//
#include <hip/hip_runtime.h>
#include <hip/hip_bf16.h>

typedef __attribute__((ext_vector_type(16))) _Float16 v16h;
typedef __attribute__((ext_vector_type(8)))  float    v8f;
typedef __attribute__((ext_vector_type(4)))  unsigned int v4u;
typedef __attribute__((ext_vector_type(8)))  int      v8i;
typedef __attribute__((ext_vector_type(4)))  int      v4i;

union FragH {
    v16h v;
    uint4 q[2];
};

// ---------------- static config ----------------
#define BATCH 8
#define CINC 32
#define COUTC 32
#define HH 256
#define WW 256
#define HU 512
#define WU 512
#define NTAPS 49           // 7x7
#define CPAD 40            // padded cin in LDS (80B stride, 16B aligned, conflict-free)
#define NPLANE (HH * WW)   // 65536
#define NROWS 9            // up-rows staged per block (2 output rows)
#define MW 68              // per-parity x entries staged
#define PAR_BYTES (NROWS * MW * CPAD * 2)   // 48960 B per parity half

// workspace layout (bytes)
#define WS_APACK 0ull                              // 49*2*32*16 f16 = 100352 B
#define WS_UP    100352ull                         // 8*512*512*32 f16 = 134217728 B
#define WS_PART  (100352ull + 134217728ull)        // 16384*32 float2 = 4194304 B
#define WS_STATS (WS_PART + 4194304ull)            // 8*32 float2

// ---------------------------------------------------------------------------
// Pass 0: assemble dense DISCO kernel from basis, pack into WMMA A-fragments.
// A-fragment layout (16-bit A 16x32, CDNA5 ISA):
//   lane<16 : elem j -> K = (j<8 ? j   : j+8)
//   lane>=16: elem j -> K = (j<8 ? j+8 : j+16),  row M = lane%16
// ---------------------------------------------------------------------------
__global__ __launch_bounds__(256) void disco_pack_weights(const float* __restrict__ weight,
                                                          _Float16* __restrict__ apack) {
    __shared__ float psiS[36 * 49];
    const float dg    = 2.0f / 512.0f;
    const float rc    = 0.015f;
    const float drr   = rc / 6.0f;
    const float PI_F  = 3.14159265358979323846f;
    const float PI2   = 6.28318530717958647692f;
    const float dphi  = PI2 / 7.0f;
    const float rnorm = rc * 6.0f / 7.0f;
    const float qn    = (dg * dg) / (PI_F * rnorm * rnorm);

    for (int idx = threadIdx.x; idx < 36 * 49; idx += 256) {
        int k = idx / 49, t = idx - k * 49;
        int ky = t / 7, kx = t - ky * 7;
        float oy = (float)(ky - 3) * dg;
        float ox = (float)(kx - 3) * dg;
        float r   = sqrtf(ox * ox + oy * oy);
        float phi = atan2f(oy, ox);
        if (phi < 0.0f) phi += PI2;
        float ir = (k == 0) ? 0.0f : (float)((k - 1) / 7 + 1) * drr;
        float rv = 1.0f - fabsf(r - ir) / drr;
        rv = rv > 0.0f ? rv : 0.0f;
        if (r > rc) rv = 0.0f;
        float pv = 1.0f;
        if (k > 0) {
            float iphi = (float)((k - 1) % 7) * dphi;
            float da = fabsf(phi - iphi);
            float dm = fminf(da, PI2 - da);
            pv = 1.0f - dm / dphi;
            pv = pv > 0.0f ? pv : 0.0f;
        }
        psiS[idx] = rv * pv * qn;
    }
    __syncthreads();

    for (int pr = threadIdx.x; pr < COUTC * CINC; pr += 256) {
        int o = pr >> 5, i = pr & 31;
        float wd[NTAPS];
#pragma unroll
        for (int t = 0; t < NTAPS; ++t) wd[t] = 0.0f;
        for (int k = 0; k < 36; ++k) {
            float wk = weight[(o * CINC + i) * 36 + k];
#pragma unroll
            for (int t = 0; t < NTAPS; ++t) wd[t] += wk * psiS[k * 49 + t];
        }
        int mt = o >> 4, lo = o & 15;
        int lane, j;
        if (i < 8)       { lane = lo;      j = i;      }
        else if (i < 16) { lane = lo + 16; j = i - 8;  }
        else if (i < 24) { lane = lo;      j = i - 8;  }
        else             { lane = lo + 16; j = i - 16; }
#pragma unroll
        for (int t = 0; t < NTAPS; ++t)
            apack[(((size_t)t * 2 + mt) * 32 + lane) * 16 + j] = (_Float16)wd[t];
    }
}

// ---------------------------------------------------------------------------
// Pass 1: 2x bilinear upsample (align_corners=True) -> f16, [b][row][x][cin]
// ---------------------------------------------------------------------------
__global__ __launch_bounds__(256) void disco_upsample(const float* __restrict__ img,
                                                      _Float16* __restrict__ up) {
    size_t g = (size_t)blockIdx.x * 256 + threadIdx.x;  // 8*512*512 threads
    int x = (int)(g & 511);
    int r = (int)((g >> 9) & 511);
    int b = (int)(g >> 18);

    const float S = 255.0f / 511.0f;
    float ys = (float)r * S;
    int y0 = (int)ys; if (y0 > 254) y0 = 254;
    float wy = ys - (float)y0;
    float xs = (float)x * S;
    int x0 = (int)xs; if (x0 > 254) x0 = 254;
    float wx = xs - (float)x0;

    __align__(16) _Float16 buf[CINC];
    const float* base = img + (((size_t)b * CINC) * 256 + y0) * 256 + x0;
#pragma unroll
    for (int c = 0; c < CINC; ++c) {
        const float* p = base + (size_t)c * 65536;
        float v00 = p[0], v01 = p[1], v10 = p[256], v11 = p[257];
        float top = v00 * (1.0f - wx) + v01 * wx;
        float bot = v10 * (1.0f - wx) + v11 * wx;
        buf[c] = (_Float16)(top * (1.0f - wy) + bot * wy);
    }
    uint4* dst = (uint4*)(up + g * CINC);
    const uint4* s4 = (const uint4*)buf;
    dst[0] = s4[0]; dst[1] = s4[1]; dst[2] = s4[2]; dst[3] = s4[3];
}

// ---------------------------------------------------------------------------
// Pass 2: implicit-GEMM stride-2 7x7 conv via WMMA f32<=f16.
// Block = 4 waves = 2 output rows x 64 x-positions x 32 out channels.
// Wave = 1 row x 32 positions (2 N-tiles): 4 WMMAs per A-fragment pair.
// LDS window: 9 up-rows, x-parity split, cin padded to 40 (80B stride).
// Interior blocks stage via Tensor Data Mover (2 descriptors, HW LDS padding);
// boundary blocks use a manual zero-padding copy.
// ---------------------------------------------------------------------------
__global__ __launch_bounds__(128) void disco_conv_wmma(const _Float16* __restrict__ up,
                                                       const _Float16* __restrict__ apack,
                                                       float* __restrict__ out,
                                                       float2* __restrict__ part) {
    // lds[parity][row 0..8][m 0..67][cin 0..CPAD)
    __shared__ _Float16 lds[2 * NROWS * MW * CPAD];

    int bi = blockIdx.x;
    int xtile = bi & 3;
    int ypair = (bi >> 2) & 127;
    int bb = bi >> 9;
    int x0 = xtile * 64;
    int y0 = ypair * 2;

    int tid = threadIdx.x;
    int wave = tid >> 5;
    int lane = tid & 31;
    int ln = lane & 15;
    int hi = lane >> 4;

    int ru0 = 2 * y0 - 3;              // first staged up-row
    int xu0 = 2 * x0 - 3;              // first staged up-col (window idx 0)

    bool interior = (ypair >= 1) && (ypair <= 126) && (xtile == 1 || xtile == 2);

    if (interior) {
        if (wave == 0) {
            unsigned ldsbase = (unsigned)(size_t)(&lds[0]);
#pragma unroll
            for (int par = 0; par < 2; ++par) {
                unsigned long long ga = (unsigned long long)(size_t)(
                    up + (((size_t)bb * HU + ru0) * WU + (xu0 + par)) * CINC);
                v4u g0;
                g0.x = 1u;                                   // count=1 (valid), user mode
                g0.y = ldsbase + (unsigned)(par * PAR_BYTES);// lds_addr
                g0.z = (unsigned)(ga & 0xffffffffull);       // global_addr[31:0]
                g0.w = (unsigned)((ga >> 32) & 0x1ffffffull) | (2u << 30); // [56:32] | type=2
                v8i g1;
                g1[0] = (1 << 16)      // data_size = 2B
                      | (1 << 20)      // pad_enable
                      | (3 << 22)      // pad_interval: 16 dwords (64B chunk)
                      | (3 << 25);     // pad_amount: 4 dwords (16B -> 80B stride)
                g1[1] = (32 << 16);    // tensor_dim0 = 32 (cin)
                g1[2] = (MW << 16);    // tensor_dim1 = 68 (x entries, this parity)
                g1[3] = (32 << 16);    // tile_dim0 = 32
                g1[4] = MW | (NROWS << 16);  // tile_dim1 = 68, tile_dim2 = 9
                g1[5] = 64;            // tensor_dim0_stride = 64 elems (2 x-positions)
                g1[6] = (16384 << 16); // tensor_dim1_stride = 512*32 elems (one up-row)
                g1[7] = 0;
                v4i g2;
                g2[0] = NROWS;         // tensor_dim2 = 9
                g2[1] = 0; g2[2] = 0; g2[3] = 0;
                v4i g3;
                g3[0] = 0; g3[1] = 0; g3[2] = 0; g3[3] = 0;
                v8i g4;                // extra operand in 6-arg toolchain form: zero-filled
                g4[0] = 0; g4[1] = 0; g4[2] = 0; g4[3] = 0;
                g4[4] = 0; g4[5] = 0; g4[6] = 0; g4[7] = 0;
                __builtin_amdgcn_tensor_load_to_lds(g0, g1, g2, g3, g4, 0);
            }
            __builtin_amdgcn_s_wait_tensorcnt(0);
        }
    } else {
        // manual staging with zero padding at tensor borders
        for (int c = tid; c < 2 * NROWS * MW; c += 128) {
            int par = c / (NROWS * MW);
            int rem = c - par * (NROWS * MW);
            int ky = rem / MW;
            int m = rem - ky * MW;
            int xu = xu0 + 2 * m + par;
            int ru = ru0 + ky;
            uint4 z{0u, 0u, 0u, 0u};
            uint4 d0 = z, d1 = z, d2 = z, d3 = z;
            if (xu >= 0 && xu < WU && ru >= 0 && ru < HU) {
                const uint4* src = (const uint4*)(up + (((size_t)bb * HU + ru) * WU + xu) * CINC);
                d0 = src[0]; d1 = src[1]; d2 = src[2]; d3 = src[3];
            }
            uint4* dst = (uint4*)&lds[((par * NROWS + ky) * MW + m) * CPAD];
            dst[0] = d0; dst[1] = d1; dst[2] = d2; dst[3] = d3;
        }
    }
    __syncthreads();

    // ---- 49 taps x 2 M-tiles x 2 N-tiles of V_WMMA_F32_16X16X32_F16 ----
    int dy = wave >> 1;                 // output row within pair
    int nb = (wave & 1) * 32;           // first of this wave's 32 positions
    v8f acc00 = {}, acc01 = {}, acc10 = {}, acc11 = {};
    for (int t = 0; t < NTAPS; ++t) {
        int ky = t / 7, kx = t - ky * 7;
        int par = kx & 1;
        int rowi = ky + 2 * dy;         // window row for this output row
        int mb = nb + ln + (kx >> 1);
        // B fragments: lanes<16 K=0..7,16..23, lanes>=16 K=8..15,24..31, col = lane%16
        FragH b0, b1;
        const uint4* bp0 = (const uint4*)&lds[((par * NROWS + rowi) * MW + mb) * CPAD + hi * 8];
        const uint4* bp1 = (const uint4*)&lds[((par * NROWS + rowi) * MW + mb + 16) * CPAD + hi * 8];
        b0.q[0] = bp0[0]; b0.q[1] = bp0[2];
        b1.q[0] = bp1[0]; b1.q[1] = bp1[2];
        // A fragments: pre-packed, 32B contiguous per lane, reused by both N-tiles
        FragH a0, a1;
        const uint4* a0p = (const uint4*)(apack + (((size_t)t * 2 + 0) * 32 + lane) * 16);
        const uint4* a1p = (const uint4*)(apack + (((size_t)t * 2 + 1) * 32 + lane) * 16);
        a0.q[0] = a0p[0]; a0.q[1] = a0p[1];
        a1.q[0] = a1p[0]; a1.q[1] = a1p[1];
        acc00 = __builtin_amdgcn_wmma_f32_16x16x32_f16(false, a0.v, false, b0.v,
                                                       (short)0, acc00, false, false);
        acc10 = __builtin_amdgcn_wmma_f32_16x16x32_f16(false, a1.v, false, b0.v,
                                                       (short)0, acc10, false, false);
        acc01 = __builtin_amdgcn_wmma_f32_16x16x32_f16(false, a0.v, false, b1.v,
                                                       (short)0, acc01, false, false);
        acc11 = __builtin_amdgcn_wmma_f32_16x16x32_f16(false, a1.v, false, b1.v,
                                                       (short)0, acc11, false, false);
    }

    // ---- store raw conv output (normalized later) ----
    // C layout: VGPR r -> (M = r + hi*8, N = ln)
    int yy = y0 + dy;
    int xA = x0 + nb + ln;
    int xB = xA + 16;
#pragma unroll
    for (int r = 0; r < 8; ++r) {
        int ch0 = r + hi * 8;
        int ch1 = 16 + ch0;
        size_t p0 = (((size_t)bb * COUTC + ch0) * HH + yy) * WW;
        size_t p1 = (((size_t)bb * COUTC + ch1) * HH + yy) * WW;
        out[p0 + xA] = acc00[r];
        out[p1 + xA] = acc10[r];
        out[p0 + xB] = acc01[r];
        out[p1 + xB] = acc11[r];
    }

    // ---- per-wave per-channel partial sum / sumsq (fixed order, no atomics) ----
    float2* wp = part + ((size_t)bi * 4 + wave) * COUTC;
#pragma unroll
    for (int r = 0; r < 8; ++r) {
        float s0 = acc00[r] + acc01[r];
        float q0 = acc00[r] * acc00[r] + acc01[r] * acc01[r];
        float s1 = acc10[r] + acc11[r];
        float q1 = acc10[r] * acc10[r] + acc11[r] * acc11[r];
#pragma unroll
        for (int off = 1; off < 16; off <<= 1) {  // xor<16 keeps lane halves separate
            s0 += __shfl_xor(s0, off, 32);
            q0 += __shfl_xor(q0, off, 32);
            s1 += __shfl_xor(s1, off, 32);
            q1 += __shfl_xor(q1, off, 32);
        }
        if (ln == 0) {
            wp[r + hi * 8] = make_float2(s0, q0);
            wp[16 + r + hi * 8] = make_float2(s1, q1);
        }
    }
}

// ---------------------------------------------------------------------------
// Pass 3: deterministic fixed-order reduction of partials -> mean, rsqrt(var+eps)
// ---------------------------------------------------------------------------
__global__ __launch_bounds__(256) void disco_stats(const float2* __restrict__ part,
                                                   float2* __restrict__ stats) {
    int bc = blockIdx.x;       // b*32 + c
    int b = bc >> 5, c = bc & 31;
    float s = 0.0f, q = 0.0f;
    for (int i = threadIdx.x; i < 2048; i += 256) {   // 512 blocks * 4 waves per batch
        float2 v = part[((size_t)b * 2048 + i) * COUTC + c];
        s += v.x;
        q += v.y;
    }
    __shared__ float ss[256], qs[256];
    ss[threadIdx.x] = s;
    qs[threadIdx.x] = q;
    __syncthreads();
    for (int st = 128; st > 0; st >>= 1) {
        if (threadIdx.x < st) {
            ss[threadIdx.x] += ss[threadIdx.x + st];
            qs[threadIdx.x] += qs[threadIdx.x + st];
        }
        __syncthreads();
    }
    if (threadIdx.x == 0) {
        float mean = ss[0] * (1.0f / (float)NPLANE);
        float var = qs[0] * (1.0f / (float)NPLANE) - mean * mean;
        stats[bc] = make_float2(mean, rsqrtf(var + 1e-5f));
    }
}

// ---------------------------------------------------------------------------
// Pass 4: in-place InstanceNorm + LeakyReLU(0.2), float4 vectorized
// ---------------------------------------------------------------------------
__global__ __launch_bounds__(256) void disco_norm(float* __restrict__ out,
                                                  const float2* __restrict__ stats) {
    size_t i = (size_t)blockIdx.x * 256 + threadIdx.x;  // float4 index; 16384 per plane
    float2 st = stats[i >> 14];
    float4* p = (float4*)out;
    float4 v = p[i];
    float t;
    t = (v.x - st.x) * st.y; v.x = t >= 0.0f ? t : 0.2f * t;
    t = (v.y - st.x) * st.y; v.y = t >= 0.0f ? t : 0.2f * t;
    t = (v.z - st.x) * st.y; v.z = t >= 0.0f ? t : 0.2f * t;
    t = (v.w - st.x) * st.y; v.w = t >= 0.0f ? t : 0.2f * t;
    p[i] = v;
}

// ---------------------------------------------------------------------------
extern "C" void kernel_launch(void* const* d_in, const int* in_sizes, int n_in,
                              void* d_out, int out_size, void* d_ws, size_t ws_size,
                              hipStream_t stream) {
    const float* img = (const float*)d_in[0];  // [8,32,256,256]
    const float* wgt = (const float*)d_in[1];  // [32,32,36]
    float* out = (float*)d_out;                // [8,32,256,256]
    char* ws = (char*)d_ws;

    _Float16* apack = (_Float16*)(ws + WS_APACK);
    _Float16* up    = (_Float16*)(ws + WS_UP);
    float2*   part  = (float2*)(ws + WS_PART);
    float2*   stats = (float2*)(ws + WS_STATS);

    disco_pack_weights<<<1, 256, 0, stream>>>(wgt, apack);
    disco_upsample<<<(BATCH * HU * WU) / 256, 256, 0, stream>>>(img, up);
    disco_conv_wmma<<<BATCH * (HH / 2) * (WW / 64), 128, 0, stream>>>(up, apack, out, part);
    disco_stats<<<BATCH * COUTC, 256, 0, stream>>>(part, stats);
    disco_norm<<<(BATCH * COUTC * NPLANE / 4) / 256, 256, 0, stream>>>(out, stats);
}
